// VariationalDequantizer_45707041964564
// MI455X (gfx1250) — matrix operand order repaired and checked by reading, not compiled, not run on hardware
//
#include <hip/hip_runtime.h>
#include <hip/hip_bf16.h>

// ---------------------------------------------------------------------------
// EGNN variational dequantizer, one workgroup per batch element.
// BS=128, N=64, D=3, NODE_NF=6, HID=64, NL=4.
// All GEMMs on v_wmma_f32_16x16x32_bf16 (wave32, fp32 accum).
// ---------------------------------------------------------------------------

typedef __attribute__((ext_vector_type(16))) __bf16          v16bf;
typedef __attribute__((ext_vector_type(16))) unsigned short  v16us;
typedef __attribute__((ext_vector_type(8)))  float           v8f;

#define GB  128   // batch
#define GN  64    // nodes per graph
#define HID 64
#define NLY 4
#define LOG2PI 1.8378770664093453f

// ---- dynamic LDS layout (bytes) -------------------------------------------
constexpr int OFF_H    = 0;        // f32 [64][64]
constexpr int OFF_HBF  = 16384;    // bf16[64][64]
constexpr int OFF_HA   = 24576;    // f32 [64][64]  h @ W1a
constexpr int OFF_HB   = 40960;    // f32 [64][64]  h @ W1b + eb1
constexpr int OFF_RAD  = 57344;    // f32 [64][64]  radial
constexpr int OFF_AGG  = 73728;    // f32 [64][64]  segment sum
constexpr int OFF_ABF  = 90112;    // bf16[64][64]
constexpr int OFF_TBF  = 98304;    // bf16[64][64]  node hidden
constexpr int OFF_U    = 106496;   // bf16 8 waves x [64][64] edge staging
constexpr int OFF_W1AT = 172032;   // bf16 [64][64]  (transposed: [n][k])
constexpr int OFF_W1BT = 180224;   // bf16 [64][64]
constexpr int OFF_W2T  = 188416;   // bf16 [64][64]
constexpr int OFF_WN1T = 196608;   // bf16 [64][128]
constexpr int OFF_WN2T = 212992;   // bf16 [64][64]
constexpr int OFF_W1R  = 221184;   // f32 [64]
constexpr int OFF_EB1  = 221440;   // f32 [64]
constexpr int OFF_EB2  = 221696;   // f32 [64]
constexpr int OFF_NB1  = 221952;   // f32 [64]
constexpr int OFF_NB2  = 222208;   // f32 [64]
constexpr int OFF_NM   = 222464;   // f32 [64]
constexpr int OFF_X    = 222720;   // f32 [64][3]
constexpr int OFF_OUT  = 223488;   // f32 [64][12]
constexpr int OFF_RED  = 226560;   // f32 [256]
constexpr int SMEM_BYTES = 227584; // ~222 KB < 320 KB/WGP

__device__ __forceinline__ unsigned short f2bf(float f) {  // RNE f32->bf16
    unsigned int u = __builtin_bit_cast(unsigned int, f);
    u += 0x7fffu + ((u >> 16) & 1u);
    return (unsigned short)(u >> 16);
}
__device__ __forceinline__ float fast_rcp(float x) { return __builtin_amdgcn_rcpf(x); }
__device__ __forceinline__ float silu(float x) {           // x * sigmoid(x), v_rcp
    return x * fast_rcp(1.f + __expf(-x));
}

// A-matrix tile 16x32 bf16 (row-major source, per ISA 7.12.2 layout):
// lane half=lane>>4, M=lane&15; k = (idx&7) + (idx>>3)*16 + half*8
__device__ __forceinline__ v16bf lds_load_a16(const unsigned short* base,
                                              int row0, int col0, int stride, int lane) {
    const int half = (lane >> 4) & 1;
    const int m    = lane & 15;
    const unsigned short* p = base + (row0 + m) * stride + col0 + half * 8;
    v16us v;
#pragma unroll
    for (int q = 0; q < 8; ++q) v[q] = p[q];          // K chunk 0..7  (+half*8)
#pragma unroll
    for (int q = 0; q < 8; ++q) v[q + 8] = p[q + 16]; // K chunk 16..23(+half*8)
    return __builtin_bit_cast(v16bf, v);
}

// B-matrix tile 32x16 bf16 from TRANSPOSED weight wT[n][k] (stride=K):
// lane half=lane>>4, N=lane&15; k = (idx&7) + (idx>>3)*16 + half*8
__device__ __forceinline__ v16bf lds_load_b16(const unsigned short* wT,
                                              int n0, int k0, int stride, int lane) {
    const int grp = (lane >> 4) & 1;
    const int n   = lane & 15;
    const unsigned short* p = wT + (n0 + n) * stride + k0 + grp * 8;
    v16us v;
#pragma unroll
    for (int q = 0; q < 8; ++q) v[q] = p[q];
#pragma unroll
    for (int q = 0; q < 8; ++q) v[q + 8] = p[q + 16];
    return __builtin_bit_cast(v16bf, v);
}

__device__ __forceinline__ v8f wmma_bf16(v16bf a, v16bf b, v8f c) {
    return __builtin_amdgcn_wmma_f32_16x16x32_bf16(false, a, false, b,
                                                   (short)0, c, false, false);
}

__global__ __launch_bounds__(256, 1)
void egnn_vdq_kernel(const float* __restrict__ cat_in, const float* __restrict__ int_in,
                     const float* __restrict__ x_in,   const float* __restrict__ nm_in,
                     const float* __restrict__ eps_in,
                     const float* __restrict__ emb_w,  const float* __restrict__ emb_b,
                     const float* __restrict__ ew1,    const float* __restrict__ eb1,
                     const float* __restrict__ ew2,    const float* __restrict__ eb2,
                     const float* __restrict__ nw1,    const float* __restrict__ nb1,
                     const float* __restrict__ nw2,    const float* __restrict__ nb2,
                     const float* __restrict__ out_w,  const float* __restrict__ out_b,
                     float* __restrict__ out) {
    extern __shared__ char smem[];
    float*          s_h    = (float*)(smem + OFF_H);
    unsigned short* s_hbf  = (unsigned short*)(smem + OFF_HBF);
    float*          s_HA   = (float*)(smem + OFF_HA);
    float*          s_HB   = (float*)(smem + OFF_HB);
    float*          s_rad  = (float*)(smem + OFF_RAD);
    float*          s_agg  = (float*)(smem + OFF_AGG);
    unsigned short* s_abf  = (unsigned short*)(smem + OFF_ABF);
    unsigned short* s_tbf  = (unsigned short*)(smem + OFF_TBF);
    unsigned short* s_u    = (unsigned short*)(smem + OFF_U);
    unsigned short* s_W1aT = (unsigned short*)(smem + OFF_W1AT);
    unsigned short* s_W1bT = (unsigned short*)(smem + OFF_W1BT);
    unsigned short* s_W2T  = (unsigned short*)(smem + OFF_W2T);
    unsigned short* s_Wn1T = (unsigned short*)(smem + OFF_WN1T);
    unsigned short* s_Wn2T = (unsigned short*)(smem + OFF_WN2T);
    float*          s_w1r  = (float*)(smem + OFF_W1R);
    float*          s_eb1  = (float*)(smem + OFF_EB1);
    float*          s_eb2  = (float*)(smem + OFF_EB2);
    float*          s_nb1  = (float*)(smem + OFF_NB1);
    float*          s_nb2  = (float*)(smem + OFF_NB2);
    float*          s_nm   = (float*)(smem + OFF_NM);
    float*          s_x    = (float*)(smem + OFF_X);
    float*          s_out  = (float*)(smem + OFF_OUT);
    float*          s_red  = (float*)(smem + OFF_RED);

    const int b    = blockIdx.x;
    const int tid  = threadIdx.x;
    const int lane = tid & 31;
    const int wave = tid >> 5;
    const int half = (lane >> 4) & 1;
    const int nn   = lane & 15;

    // ---------------- Phase 1: inputs, embedding, radial -------------------
    if (tid < GN)      s_nm[tid] = nm_in[(size_t)b * GN + tid];
    if (tid < GN * 3)  s_x[tid]  = x_in[(size_t)b * GN * 3 + tid];
    __syncthreads();

#pragma unroll
    for (int it = 0; it < 16; ++it) {           // h = (concat*nm) @ emb_w + emb_b
        int idx  = it * 256 + tid;
        int node = idx >> 6, f = idx & 63;
        float nmv = s_nm[node];
        const float* cp = cat_in + ((size_t)b * GN + node) * 5;
        float iv  = int_in[(size_t)b * GN + node];
        float acc = emb_b[f];
        acc += cp[0] * nmv * emb_w[0 * 64 + f];
        acc += cp[1] * nmv * emb_w[1 * 64 + f];
        acc += cp[2] * nmv * emb_w[2 * 64 + f];
        acc += cp[3] * nmv * emb_w[3 * 64 + f];
        acc += cp[4] * nmv * emb_w[4 * 64 + f];
        acc += iv    * nmv * emb_w[5 * 64 + f];
        s_h[idx] = acc;  s_hbf[idx] = f2bf(acc);
    }
#pragma unroll
    for (int it = 0; it < 16; ++it) {           // radial[i][j] = ||x_i*nm - x_j*nm||^2
        int idx = it * 256 + tid;
        int i = idx >> 6, j = idx & 63;
        float ni = s_nm[i], nj = s_nm[j];
        float d0 = s_x[i * 3 + 0] * ni - s_x[j * 3 + 0] * nj;
        float d1 = s_x[i * 3 + 1] * ni - s_x[j * 3 + 1] * nj;
        float d2 = s_x[i * 3 + 2] * ni - s_x[j * 3 + 2] * nj;
        s_rad[idx] = d0 * d0 + d1 * d1 + d2 * d2;
    }
    __syncthreads();

    // ---------------- Layers ------------------------------------------------
    for (int l = 0; l < NLY; ++l) {
        // zero agg + stage this layer's weights (transposed, bf16)
        const float* ew1l = ew1 + (size_t)l * 129 * 64;
#pragma unroll
        for (int it = 0; it < 16; ++it) {
            int idx = it * 256 + tid;
            int n = idx >> 6, k = idx & 63;
            s_agg[idx] = 0.f;
            s_W1aT[n * 64 + k] = f2bf(ew1l[(size_t)k * 64 + n]);
            s_W1bT[n * 64 + k] = f2bf(ew1l[(size_t)(64 + k) * 64 + n]);
            s_W2T [n * 64 + k] = f2bf(ew2[((size_t)l * 64 + k) * 64 + n]);
            s_Wn2T[n * 64 + k] = f2bf(nw2[((size_t)l * 64 + k) * 64 + n]);
        }
#pragma unroll
        for (int it = 0; it < 32; ++it) {
            int idx = it * 256 + tid;
            int n = idx >> 7, k = idx & 127;
            s_Wn1T[n * 128 + k] = f2bf(nw1[((size_t)l * 128 + k) * 64 + n]);
        }
        if (tid < 64) {
            s_w1r[tid] = ew1l[128 * 64 + tid];
            s_eb1[tid] = eb1[l * 64 + tid];
            s_eb2[tid] = eb2[l * 64 + tid];
            s_nb1[tid] = nb1[l * 64 + tid];
            s_nb2[tid] = nb2[l * 64 + tid];
        }
        __syncthreads();

        // HA = h@W1a ; HB = h@W1b + eb1   (two 64x64x64 GEMMs, 2 tiles/wave)
#pragma unroll
        for (int tt = 0; tt < 2; ++tt) {
            int t = wave + tt * 8, mt = t >> 2, nt = t & 3;
            v8f accA = {0.f,0.f,0.f,0.f,0.f,0.f,0.f,0.f};
            v8f accB = {0.f,0.f,0.f,0.f,0.f,0.f,0.f,0.f};
#pragma unroll
            for (int kk = 0; kk < 2; ++kk) {
                v16bf a  = lds_load_a16(s_hbf, mt * 16, kk * 32, 64, lane);
                v16bf bA = lds_load_b16(s_W1aT, nt * 16, kk * 32, 64, lane);
                v16bf bB = lds_load_b16(s_W1bT, nt * 16, kk * 32, 64, lane);
                accA = wmma_bf16(a, bA, accA);
                accB = wmma_bf16(a, bB, accB);
            }
            int n = nt * 16 + nn;
            float ebv = s_eb1[n];
#pragma unroll
            for (int r = 0; r < 8; ++r) {
                int row = mt * 16 + r + 8 * half;
                s_HA[row * 64 + n] = accA[r];
                s_HB[row * 64 + n] = accB[r] + ebv;
            }
        }
        __syncthreads();

        // ---- Edge phase: wave owns node i ---------------------------------
        // Preload layer-invariant B tiles (W2^T) + biases into registers.
        v16bf Bt0[4], Bt1[4];
        float ebv2[4];
#pragma unroll
        for (int nt = 0; nt < 4; ++nt) {
            Bt0[nt]  = lds_load_b16(s_W2T, nt * 16, 0,  64, lane);
            Bt1[nt]  = lds_load_b16(s_W2T, nt * 16, 32, 64, lane);
            ebv2[nt] = s_eb2[nt * 16 + nn];
        }
        for (int i = wave; i < GN; i += 8) {
            float nmi = s_nm[i];
            // build u_i = silu(HA_i + HB_j + r*w1r): lane owns k=2*lane, 2*lane+1
            const int k2 = lane * 2;
            const float2 hav = *(const float2*)(s_HA + i * 64 + k2);
            const float2 wrv = *(const float2*)(s_w1r + k2);
            unsigned short* uw = s_u + wave * 4096;
            unsigned int*   uw32 = (unsigned int*)uw;
            for (int j = 0; j < GN; ++j) {
                float rv = s_rad[i * 64 + j];
                const float2 hbv = *(const float2*)(s_HB + j * 64 + k2);
                float v0 = silu(hav.x + hbv.x + rv * wrv.x);
                float v1 = silu(hav.y + hbv.y + rv * wrv.y);
                uw32[(j * 64 + k2) >> 1] =
                    (unsigned)f2bf(v0) | ((unsigned)f2bf(v1) << 16);
            }
            asm volatile("s_wait_dscnt 0x0" ::: "memory");  // staging -> own wave reads
            float colacc[4] = {0.f, 0.f, 0.f, 0.f};
#pragma unroll
            for (int mt = 0; mt < 4; ++mt) {
                v16bf a0 = lds_load_a16(uw, mt * 16, 0,  64, lane);
                v16bf a1 = lds_load_a16(uw, mt * 16, 32, 64, lane);
                float nmj[8];
#pragma unroll
                for (int r = 0; r < 8; ++r) nmj[r] = s_nm[mt * 16 + r + 8 * half];
#pragma unroll
                for (int nt = 0; nt < 4; ++nt) {
                    v8f acc = {0.f,0.f,0.f,0.f,0.f,0.f,0.f,0.f};
                    acc = wmma_bf16(a0, Bt0[nt], acc);
                    acc = wmma_bf16(a1, Bt1[nt], acc);
#pragma unroll
                    for (int r = 0; r < 8; ++r)
                        colacc[nt] += silu(acc[r] + ebv2[nt]) * nmj[r];
                }
            }
#pragma unroll
            for (int nt = 0; nt < 4; ++nt) {
                float c = colacc[nt] * nmi;
                c += __shfl_xor(c, 16, 32);
                if (lane < 16) s_agg[i * 64 + nt * 16 + lane] += c;
            }
        }
        __syncthreads();

        // agg -> bf16
#pragma unroll
        for (int it = 0; it < 16; ++it) {
            int idx = it * 256 + tid;
            s_abf[idx] = f2bf(s_agg[idx]);
        }
        __syncthreads();

        // node MLP 1: t = silu([h|agg] @ nw1 + nb1)   (64x128x64)
#pragma unroll
        for (int tt = 0; tt < 2; ++tt) {
            int t = wave + tt * 8, mt = t >> 2, nt = t & 3;
            v8f acc = {0.f,0.f,0.f,0.f,0.f,0.f,0.f,0.f};
#pragma unroll
            for (int kk = 0; kk < 4; ++kk) {
                int c0 = kk * 32;
                const unsigned short* A = (c0 < 64) ? s_hbf : s_abf;
                v16bf a  = lds_load_a16(A, mt * 16, c0 & 63, 64, lane);
                v16bf bb = lds_load_b16(s_Wn1T, nt * 16, c0, 128, lane);
                acc = wmma_bf16(a, bb, acc);
            }
            int n = nt * 16 + nn;
            float bv = s_nb1[n];
#pragma unroll
            for (int r = 0; r < 8; ++r) {
                int row = mt * 16 + r + 8 * half;
                s_tbf[row * 64 + n] = f2bf(silu(acc[r] + bv));
            }
        }
        __syncthreads();

        // node MLP 2 + residual: h += t @ nw2 + nb2
#pragma unroll
        for (int tt = 0; tt < 2; ++tt) {
            int t = wave + tt * 8, mt = t >> 2, nt = t & 3;
            v8f acc = {0.f,0.f,0.f,0.f,0.f,0.f,0.f,0.f};
#pragma unroll
            for (int kk = 0; kk < 2; ++kk) {
                v16bf a  = lds_load_a16(s_tbf,  mt * 16, kk * 32, 64, lane);
                v16bf bb = lds_load_b16(s_Wn2T, nt * 16, kk * 32, 64, lane);
                acc = wmma_bf16(a, bb, acc);
            }
            int n = nt * 16 + nn;
            float bv = s_nb2[n];
#pragma unroll
            for (int r = 0; r < 8; ++r) {
                int row = mt * 16 + r + 8 * half;
                int idx = row * 64 + n;
                float hv = s_h[idx] + acc[r] + bv;
                s_h[idx] = hv;  s_hbf[idx] = f2bf(hv);
            }
        }
        __syncthreads();
    }

    // ---------------- Output projection (64x12, VALU) -----------------------
    for (int idx = tid; idx < GN * 12; idx += 256) {
        int n = idx / 12, o = idx % 12;
        float acc = out_b[o];
#pragma unroll
        for (int k = 0; k < 64; ++k) acc += s_h[n * 64 + k] * out_w[k * 12 + o];
        s_out[idx] = acc * s_nm[n];   // net_out * nm_flat
    }
    __syncthreads();

    // ---------------- Flow tail + per-batch log_qv --------------------------
    float part = 0.f;
    for (int idx = tid; idx < GN * 6; idx += 256) {
        int n = idx / 6, f = idx % 6;
        float nmv = s_nm[n];
        float mu = s_out[n * 12 + f];
        float ls = s_out[n * 12 + 6 + f];
        float ep = eps_in[((size_t)b * GN + n) * 6 + f] * nmv;
        part += nmv * (-0.5f * ep * ep - 0.5f * LOG2PI) - ls;    // log_q_eps - sum(log_sigma)
        float uu = mu + ep * __expf(ls);
        float sg = fast_rcp(1.f + __expf(-uu));
        float au = fabsf(uu);
        float ldj = -au - 2.f * log1pf(__expf(-au));             // logsig(u)+logsig(-u)
        part -= nmv * ldj;
        float base = (f < 5) ? cat_in[((size_t)b * GN + n) * 5 + f]
                             : int_in[(size_t)b * GN + n];
        float v = (base + sg) * nmv;
        if (f < 5) out[((size_t)b * GN + n) * 5 + f] = v;        // v_cat
        else       out[(size_t)GB * GN * 5 + (size_t)b * GN + n] = v; // v_int
    }
    s_red[tid] = part;
    __syncthreads();
#pragma unroll
    for (int s = 128; s > 0; s >>= 1) {
        if (tid < s) s_red[tid] += s_red[tid + s];
        __syncthreads();
    }
    if (tid == 0) out[(size_t)GB * GN * 6 + b] = s_red[0];       // log_qv
}

extern "C" void kernel_launch(void* const* d_in, const int* in_sizes, int n_in,
                              void* d_out, int out_size, void* d_ws, size_t ws_size,
                              hipStream_t stream) {
    (void)in_sizes; (void)n_in; (void)out_size; (void)d_ws; (void)ws_size;
    const float* cat_in = (const float*)d_in[0];
    const float* int_in = (const float*)d_in[1];
    const float* x_in   = (const float*)d_in[2];
    const float* nm_in  = (const float*)d_in[3];
    const float* eps_in = (const float*)d_in[4];
    // d_in[5] = rows, d_in[6] = cols : dense per-batch edge structure, unused
    const float* emb_w  = (const float*)d_in[7];
    const float* emb_b  = (const float*)d_in[8];
    const float* ew1    = (const float*)d_in[9];
    const float* eb1    = (const float*)d_in[10];
    const float* ew2    = (const float*)d_in[11];
    const float* eb2    = (const float*)d_in[12];
    const float* nw1    = (const float*)d_in[13];
    const float* nb1    = (const float*)d_in[14];
    const float* nw2    = (const float*)d_in[15];
    const float* nb2    = (const float*)d_in[16];
    const float* out_w  = (const float*)d_in[17];
    const float* out_b  = (const float*)d_in[18];
    float* out = (float*)d_out;

    egnn_vdq_kernel<<<dim3(GB), dim3(256), SMEM_BYTES, stream>>>(
        cat_in, int_in, x_in, nm_in, eps_in,
        emb_w, emb_b, ew1, eb1, ew2, eb2, nw1, nb1, nw2, nb2,
        out_w, out_b, out);
}
